// Model_53163105190192
// MI455X (gfx1250) — matrix operand order repaired and checked by reading, not compiled, not run on hardware
//
#include <hip/hip_runtime.h>
#include <hip/hip_bf16.h>

// Problem constants (match reference)
#define DD 128
#define LL 8
#define MM 32768
#define NN (LL * MM)
#define GG 5
#define FF 2
#define NPB 16            // nodes per block in level kernel
#define AROWS (NPB * FF)  // 32 A-matrix rows per block

typedef __attribute__((ext_vector_type(16))) __bf16 v16bf;
typedef __attribute__((ext_vector_type(8)))  float  v8f;

// Pre-converted weight arena layout (bf16, elements) inside d_ws:
//   [0, 32768)                      hs_W^T   [n*256+k]
//   gate g block at 32768 + g*114688:
//     +0      W1^T (128n x 256k)  32768 elems
//     +32768  W2^T (128n x 128k)  16384 elems
//     +49152  W3^T (128n x 128k)  16384 elems
//     +65536  Wi^T (384n x 128k)  49152 elems
#define WPG 114688
#define WHS 32768
#define WTOTAL (WHS + GG * WPG)

__device__ __forceinline__ unsigned short f2bf(float f) {
  unsigned u = __builtin_bit_cast(unsigned, f);
  u += 0x7FFFu + ((u >> 16) & 1u);   // round-to-nearest-even
  return (unsigned short)(u >> 16);
}
__device__ __forceinline__ float bf2f(unsigned short h) {
  return __builtin_bit_cast(float, ((unsigned)h) << 16);
}
__device__ __forceinline__ float sigmoidf(float x) { return 1.0f / (1.0f + __expf(-x)); }

// ---------------- async global -> LDS staging (CDNA5 ASYNCcnt path) ----------------
#if __has_builtin(__builtin_amdgcn_global_load_async_to_lds_b128) && \
    __has_builtin(__builtin_amdgcn_s_wait_asynccnt)
#define HAVE_ASYNC_LDS 1
typedef __attribute__((ext_vector_type(4))) int v4i;
typedef __attribute__((address_space(1))) v4i gv4i;   // global 16B vector
typedef __attribute__((address_space(3))) v4i lv4i;   // LDS 16B vector

__device__ __forceinline__ gv4i* to_global_v4(const void* p) {
  return (gv4i*)(unsigned long long)(size_t)p;        // inttoptr: flat == global
}
__device__ __forceinline__ lv4i* to_lds_v4(void* p) {
  return (lv4i*)(unsigned)(size_t)p;                  // LDS offset = addr[31:0]
}
#else
#define HAVE_ASYNC_LDS 0
#endif

// nbytes must be a multiple of nthreads*16 (true for all call sites here).
__device__ __forceinline__ void stage_lds(unsigned short* dst_lds,
                                          const unsigned short* src_g,
                                          int nbytes, int t, int nthreads) {
#if HAVE_ASYNC_LDS
  for (int i = t * 16; i < nbytes; i += nthreads * 16) {
    __builtin_amdgcn_global_load_async_to_lds_b128(
        to_global_v4((const char*)src_g + i),
        to_lds_v4((char*)dst_lds + i), 0, 0);
  }
  __builtin_amdgcn_s_wait_asynccnt(0);
#else
  const int4* s4 = (const int4*)src_g;
  int4* d4 = (int4*)dst_lds;
  int n4 = nbytes >> 4;
  for (int i = t; i < n4; i += nthreads) d4[i] = s4[i];
#endif
}

// ---- WMMA fragment loaders (wave32, bf16 16x16x32, layouts per CDNA5 ISA 7.12.2) ----

// A fragment: 16x32 (MxK). lanes 0-15: M=lane, K base 0; lanes 16-31: M=lane-16, +8.
__device__ __forceinline__ v16bf load_fragA(const unsigned short* A, int Ast,
                                            int rowbase, int kc, int lane) {
  union { unsigned int w[8]; v16bf v; } u;
  const unsigned short* base =
      A + (rowbase + (lane & 15)) * Ast + kc + ((lane >> 4) * 8);
#pragma unroll
  for (int v = 0; v < 8; ++v) {
    int kb = ((v & 3) * 2) + ((v >> 2) * 16);
    u.w[v] = *(const unsigned int*)(base + kb);
  }
  return u.v;
}

// B fragment: 32x16 (KxN) from K-transposed LDS image Bt[n*Bst + k].
__device__ __forceinline__ v16bf load_fragB(const unsigned short* Bt, int Bst,
                                            int colbase, int kc, int lane) {
  union { unsigned int w[8]; v16bf v; } u;
  const unsigned short* base =
      Bt + (colbase + (lane & 15)) * Bst + kc + ((lane >> 4) * 8);
#pragma unroll
  for (int v = 0; v < 8; ++v) u.w[v] = *(const unsigned int*)(base + ((v & 3) * 2) + ((v >> 2) * 16));
  return u.v;
}

__device__ __forceinline__ v8f wmma_bf16(v16bf a, v16bf b, v8f c) {
  return __builtin_amdgcn_wmma_f32_16x16x32_bf16(false, a, false, b, (short)0, c,
                                                 false, false);
}

// C/D layout: c[r] -> (m = rowbase + (lane>>4)*8 + r, n = colbase + (lane&15))
__device__ __forceinline__ void gemm_lds(const unsigned short* A, int Ast, int K,
                                         const unsigned short* Bt, int Bst,
                                         unsigned short* Out, int Ost,
                                         const float* bias, bool do_relu,
                                         int rowTiles, int colTiles,
                                         int wave, int lane) {
  int T = rowTiles * colTiles;
  for (int tt = wave; tt < T; tt += 4) {
    int rt = tt / colTiles, ct = tt % colTiles;
    v8f acc = {};
    for (int kc = 0; kc < K; kc += 32) {
      v16bf a = load_fragA(A, Ast, rt * 16, kc, lane);
      v16bf b = load_fragB(Bt, Bst, ct * 16, kc, lane);
      acc = wmma_bf16(a, b, acc);
    }
    int d = ct * 16 + (lane & 15);
    float bv = bias[d];
    int mbase = rt * 16 + ((lane >> 4) * 8);
#pragma unroll
    for (int r = 0; r < 8; ++r) {
      float v = acc[r] + bv;
      if (do_relu) v = fmaxf(v, 0.0f);
      Out[(mbase + r) * Ost + d] = f2bf(v);
    }
  }
}

// ---------------- Weight pre-convert/transpose (once per call) ----------------
__global__ void prep_weights(const float* __restrict__ hsW,
                             const float* __restrict__ W1,
                             const float* __restrict__ W2,
                             const float* __restrict__ W3,
                             const float* __restrict__ Wi,
                             unsigned short* __restrict__ out) {
  int e = blockIdx.x * blockDim.x + threadIdx.x;
  if (e >= WTOTAL) return;
  if (e < WHS) {                       // hs_W^T: [n*256+k] from hsW[k*128+n]
    int n = e >> 8, k = e & 255;
    out[e] = f2bf(hsW[k * DD + n]);
    return;
  }
  int r = e - WHS;
  int g = r / WPG; r %= WPG;
  unsigned short* og = out + WHS + (size_t)g * WPG;
  if (r < 32768) {                     // W1^T
    int n = r >> 8, k = r & 255;
    og[r] = f2bf(W1[((size_t)g * 256 + k) * DD + n]);
  } else if (r < 49152) {              // W2^T
    int rr = r - 32768; int n = rr >> 7, k = rr & 127;
    og[r] = f2bf(W2[((size_t)g * DD + k) * DD + n]);
  } else if (r < 65536) {              // W3^T
    int rr = r - 49152; int n = rr >> 7, k = rr & 127;
    og[r] = f2bf(W3[((size_t)g * DD + k) * DD + n]);
  } else {                             // Wi^T (384 x 128)
    int rr = r - 65536; int n = rr >> 7, k = rr & 127;
    og[r] = f2bf(Wi[((size_t)g * DD + k) * (3 * DD) + n]);
  }
}

// ---------------- Kernel 1: hs = [s|t] @ hs_W + hs_b -----------------------------
__global__ void __launch_bounds__(128) hs_kernel(const float* __restrict__ s,
                                                 const float* __restrict__ tin,
                                                 const unsigned short* __restrict__ wsW,
                                                 const float* __restrict__ b,
                                                 float* __restrict__ hs) {
  extern __shared__ char smem[];
  unsigned short* Abuf = (unsigned short*)smem;                  // 64 x 256
  unsigned short* Wbuf = (unsigned short*)(smem + 64 * 256 * 2); // 128 x 256 (^T)
  int t = threadIdx.x, lane = t & 31, wave = t >> 5;
  size_t rowbase = (size_t)blockIdx.x * 64;

  stage_lds(Wbuf, wsW, WHS * 2, t, 128);        // async hs_W^T -> LDS

  for (int i = t; i < 64 * 256; i += 128) {
    int r = i >> 8, c = i & 255;
    size_t row = rowbase + r;
    float v = (c < DD) ? s[row * DD + c] : tin[row * DD + (c - DD)];
    Abuf[i] = f2bf(v);
  }
  __syncthreads();

  for (int tt = wave; tt < 32; tt += 4) {   // 4 rowTiles x 8 colTiles
    int rt = tt >> 3, ct = tt & 7;
    v8f acc = {};
    for (int kc = 0; kc < 256; kc += 32) {
      v16bf a = load_fragA(Abuf, 256, rt * 16, kc, lane);
      v16bf bb = load_fragB(Wbuf, 256, ct * 16, kc, lane);
      acc = wmma_bf16(a, bb, acc);
    }
    int d = ct * 16 + (lane & 15);
    float bv = b[d];
    int mb = rt * 16 + ((lane >> 4) * 8);
#pragma unroll
    for (int r = 0; r < 8; ++r)
      hs[(rowbase + mb + r) * DD + d] = acc[r] + bv;
  }
}

// ---------------- Helpers: zero / bucket-by-gate ----------------
__global__ void zero_f32(float* p, int n) {
  int i = blockIdx.x * blockDim.x + threadIdx.x;
  if (i < n) p[i] = 0.0f;
}
__global__ void zero_counts(int* cnt) {
  if (threadIdx.x < GG) cnt[threadIdx.x] = 0;
}
__global__ void bucket_kernel(const int* __restrict__ gate_lvl, int* cnt, int* idx) {
  int m = blockIdx.x * blockDim.x + threadIdx.x;
  if (m < MM) {
    int g = gate_lvl[m];
    int p = atomicAdd(&cnt[g], 1);
    idx[g * MM + p] = m;
  }
}

// ---------------- Kernel 2: fused per-level aggr MLP + GRU ------------------------
#define SM_NODES 0
#define SM_SRC   64
#define SM_A     256
#define SM_H1    (SM_A + AROWS * 256 * 2)     // 16640
#define SM_H2    (SM_H1 + AROWS * DD * 2)     // 24832
#define SM_W     (SM_H2 + AROWS * DD * 2)     // 33024
#define SM_TOTAL (SM_W + 384 * DD * 2)        // 131328 bytes

__global__ void __launch_bounds__(128) level_kernel(
    int lvl, const int* __restrict__ fanin,
    const float* __restrict__ hs, float* __restrict__ hf,
    const unsigned short* __restrict__ wsW,
    const float* __restrict__ b1, const float* __restrict__ b2,
    const float* __restrict__ b3,
    const float* __restrict__ bi, const float* __restrict__ bh,
    const int* __restrict__ cnt, const int* __restrict__ idx) {
  extern __shared__ char smem[];
  int* nodes = (int*)(smem + SM_NODES);
  int* srcRows = (int*)(smem + SM_SRC);
  unsigned short* Abuf = (unsigned short*)(smem + SM_A);
  unsigned short* H1 = (unsigned short*)(smem + SM_H1);
  unsigned short* H2 = (unsigned short*)(smem + SM_H2);
  unsigned short* Wbuf = (unsigned short*)(smem + SM_W);

  const int blocksPerGate = MM / NPB;  // 2048
  int g = blockIdx.x / blocksPerGate;
  int j = blockIdx.x % blocksPerGate;
  int c = cnt[g];
  if (j * NPB >= c) return;

  int t = threadIdx.x, lane = t & 31, wave = t >> 5;
  const unsigned short* wg = wsW + WHS + (size_t)g * WPG;

  if (t < NPB) {
    int pos = j * NPB + t;
    if (pos >= c) pos = c - 1;           // clamp tail (benign duplicate work)
    nodes[t] = idx[g * MM + pos];
  }
  __syncthreads();
  if (t < AROWS) {
    int nd = nodes[t >> 1];
    int f = t & 1;
    int src = fanin[(((lvl - 1) * MM) + nd) * FF + f];
    srcRows[t] = (lvl - 1) * MM + src;
  }
  stage_lds(Wbuf, wg, 32768 * 2, t, 128);   // async W1^T -> LDS
  __syncthreads();

  // Gather src_state = [hs[src] | hf[src]] -> A (32 x 256) bf16. 4 threads/row.
  {
    int row = t >> 2, q = t & 3;
    size_t src = (size_t)srcRows[row];
    const float* base0 = hs + src * DD;
    const float* base1 = hf + src * DD;
#pragma unroll
    for (int cc = 0; cc < 64; cc += 4) {
      int col = q * 64 + cc;
      const float* p = (col < DD) ? (base0 + col) : (base1 + (col - DD));
      float4 v = *(const float4*)p;
      unsigned short* o = Abuf + row * 256 + col;
      o[0] = f2bf(v.x); o[1] = f2bf(v.y); o[2] = f2bf(v.z); o[3] = f2bf(v.w);
    }
  }
  __syncthreads();

  // h1 = relu(A @ W1 + b1) : 32x128
  gemm_lds(Abuf, 256, 256, Wbuf, 256, H1, DD, b1 + g * DD, true, 2, 8, wave, lane);
  __syncthreads();

  stage_lds(Wbuf, wg + 32768, 16384 * 2, t, 128);   // async W2^T
  __syncthreads();
  // h2 = relu(h1 @ W2 + b2) : 32x128
  gemm_lds(H1, DD, DD, Wbuf, DD, H2, DD, b2 + g * DD, true, 2, 8, wave, lane);
  __syncthreads();

  stage_lds(Wbuf, wg + 49152, 16384 * 2, t, 128);   // async W3^T
  __syncthreads();
  // m3 = h2 @ W3 + b3 : 32x128 (into H1, h1 dead)
  gemm_lds(H2, DD, DD, Wbuf, DD, H1, DD, b3 + g * DD, false, 2, 8, wave, lane);
  __syncthreads();

  // msg[i] = m3[2i] + m3[2i+1] : 16x128 (into H2)
  for (int i = t; i < NPB * DD; i += 128) {
    int m = i >> 7, d = i & 127;
    float v = bf2f(H1[(2 * m) * DD + d]) + bf2f(H1[(2 * m + 1) * DD + d]);
    H2[m * DD + d] = f2bf(v);
  }
  stage_lds(Wbuf, wg + 65536, 49152 * 2, t, 128);   // async Wi^T (384x128)
  __syncthreads();

  // GRU: gi = msg @ Wi + bi ; gh == bh (h is zero at read time in the reference).
  for (int ct = wave; ct < 8; ct += 4) {   // 8 d-tiles of 16, msg rows = 16
    v8f ar = {}, az = {}, an = {};
    for (int kc = 0; kc < DD; kc += 32) {
      v16bf a = load_fragA(H2, DD, 0, kc, lane);
      v16bf br = load_fragB(Wbuf, DD, ct * 16, kc, lane);
      v16bf bz = load_fragB(Wbuf, DD, DD + ct * 16, kc, lane);
      v16bf bn = load_fragB(Wbuf, DD, 2 * DD + ct * 16, kc, lane);
      ar = wmma_bf16(a, br, ar);
      az = wmma_bf16(a, bz, az);
      an = wmma_bf16(a, bn, an);
    }
    int d = ct * 16 + (lane & 15);
    float bir = bi[g * 3 * DD + d];
    float biz = bi[g * 3 * DD + DD + d];
    float bin = bi[g * 3 * DD + 2 * DD + d];
    float bhr = bh[g * 3 * DD + d];
    float bhz = bh[g * 3 * DD + DD + d];
    float bhn = bh[g * 3 * DD + 2 * DD + d];
    int mbase = (lane >> 4) * 8;
#pragma unroll
    for (int r = 0; r < 8; ++r) {
      int m = mbase + r;                   // 0..15
      float rr = sigmoidf(ar[r] + bir + bhr);
      float zz = sigmoidf(az[r] + biz + bhz);
      float nn = tanhf(an[r] + bin + rr * bhn);
      float hnew = (1.0f - zz) * nn;
      size_t nd = (size_t)nodes[m];
      hf[((size_t)lvl * MM + nd) * DD + d] = hnew;
    }
  }
}

// ---------------- Launcher ----------------
extern "C" void kernel_launch(void* const* d_in, const int* in_sizes, int n_in,
                              void* d_out, int out_size, void* d_ws, size_t ws_size,
                              hipStream_t stream) {
  const float* s      = (const float*)d_in[0];
  const float* tin    = (const float*)d_in[1];
  const int*   gate   = (const int*)d_in[2];
  const int*   fanin  = (const int*)d_in[3];
  const float* hs_W   = (const float*)d_in[4];
  const float* hs_b   = (const float*)d_in[5];
  const float* aW1    = (const float*)d_in[6];
  const float* ab1    = (const float*)d_in[7];
  const float* aW2    = (const float*)d_in[8];
  const float* ab2    = (const float*)d_in[9];
  const float* aW3    = (const float*)d_in[10];
  const float* ab3    = (const float*)d_in[11];
  const float* gWi    = (const float*)d_in[12];
  // d_in[13] = gru_Wh: unused — multiplied by hf[lo:lo+M] which is always 0 at read time
  const float* gbi    = (const float*)d_in[14];
  const float* gbh    = (const float*)d_in[15];

  float* hs = (float*)d_out;                     // N x D
  float* hf = (float*)d_out + (size_t)NN * DD;   // N x D

  // Workspace carve-up: counts | per-gate index lists | bf16 transposed weights
  int* wsCnt = (int*)d_ws;
  int* wsIdx = (int*)d_ws + 16;
  size_t wOff = ((size_t)(64 + GG * MM * 4) + 255) & ~(size_t)255;
  unsigned short* wsW = (unsigned short*)((char*)d_ws + wOff);

  prep_weights<<<(WTOTAL + 255) / 256, 256, 0, stream>>>(hs_W, aW1, aW2, aW3, gWi, wsW);

  hs_kernel<<<NN / 64, 128, (64 * 256 + 128 * 256) * 2, stream>>>(s, tin, wsW, hs_b, hs);

  zero_f32<<<(MM * DD + 255) / 256, 256, 0, stream>>>(hf, MM * DD);  // hf level 0

  for (int lvl = 1; lvl < LL; ++lvl) {
    zero_counts<<<1, 32, 0, stream>>>(wsCnt);
    bucket_kernel<<<(MM + 255) / 256, 256, 0, stream>>>(gate + (size_t)lvl * MM,
                                                        wsCnt, wsIdx);
    level_kernel<<<GG * (MM / NPB), 128, SM_TOTAL, stream>>>(
        lvl, fanin, hs, hf, wsW, ab1, ab2, ab3, gbi, gbh, wsCnt, wsIdx);
  }
}
